// MultiLstm_81286551044510
// MI455X (gfx1250) — matrix operand order repaired and checked by reading, not compiled, not run on hardware
//
#include <hip/hip_runtime.h>
#include <stdint.h>

#define SEQ   2048
#define BATCH 64
#define INDIM 64
#define HID   128
#define GATES 512
#define HPITCH 136   // bf16 pitch for LDS h buffer (272B rows, 16B aligned, bank-staggered)

typedef __attribute__((ext_vector_type(16))) __bf16 v16bf;
typedef __attribute__((ext_vector_type(8)))  float  v8f;
typedef __attribute__((ext_vector_type(4)))  unsigned int u32x4;

union FragU {
  v16bf    v;
  u32x4    q[2];
  uint16_t s[16];
};

__device__ inline uint16_t f2bf(float f) {
  union { float f; uint32_t u; } x; x.f = f;
  uint32_t u = x.u;
  return (uint16_t)((u + 0x7FFFu + ((u >> 16) & 1u)) >> 16);  // RNE
}

__device__ inline v8f wmma_bf16(v16bf a, v16bf b, v8f c) {
  // D = A(16x32 bf16) * B(32x16 bf16) + C(16x16 f32)
  return __builtin_amdgcn_wmma_f32_16x16x32_bf16(false, a, false, b, (short)0, c, false, false);
}

// Fast transcendentals: raw v_rcp_f32 (~1 ulp, far below bf16 noise floor).
__device__ inline float rcp_(float x) { return __builtin_amdgcn_rcpf(x); }
__device__ inline float sigm(float x) {
  x = fminf(fmaxf(x, -30.f), 30.f);
  return rcp_(1.f + __expf(-x));
}
__device__ inline float tanh_(float x) {
  x = fminf(fmaxf(x, -15.f), 15.f);
  return fmaf(-2.f, rcp_(__expf(2.f * x) + 1.f), 1.f);
}

// Load a 32-byte fragment at a compile-time element offset from a per-lane
// base pointer: lowers to two global_load_b128 with immediate offsets.
__device__ inline v16bf load_frag_ofs(const uint16_t* base, int ofs) {
  FragU f;
  const u32x4* p = (const u32x4*)(base + ofs);
  f.q[0] = p[0];
  f.q[1] = p[1];
  return f.v;
}

// A fragment: two 16-byte runs at ofs and ofs+16 (K = kb+0..7, kb+16..23).
__device__ inline v16bf load_a_ofs(const uint16_t* base, int ofs) {
  FragU f;
  f.q[0] = *(const u32x4*)(base + ofs);
  f.q[1] = *(const u32x4*)(base + ofs + 16);
  return f.v;
}

// ---------------------------------------------------------------------------
// Persistent LSTM layer body: one 256-thread block (8 wave32 waves).
// Wave w = hidden-column block jt (0..7). Each wave owns ALL 4 batch tiles and
// ALL 4 gates for its 16 hidden columns: 16 accumulator tiles (>256 VGPRs via
// the gfx1250 VGPR-MSB mechanism, 2 waves/SIMD). Every weight B-fragment is
// loaded once per step (L2-resident) and reused across 4 WMMAs; A fragments
// come from LDS. c stays in VGPRs the entire sequence.
// LAYER==0: producer — writes bf16 h-trace, publishes progress (release).
// LAYER==1: consumer — spins on progress (acquire), fuses the final linear.
// ---------------------------------------------------------------------------
template <int KIN, int LAYER>
__device__ void lstm_layer_body(
    const uint16_t* __restrict__ xbf,    // [SEQ][B][KIN] bf16 input stream
    const uint16_t* __restrict__ Wih,    // [GATES][KIN]  bf16
    const uint16_t* __restrict__ Whh,    // [GATES][HID]  bf16
    const float*    __restrict__ bias,   // [GATES] = bih+bhh
    uint16_t*       __restrict__ hout,   // LAYER==0: h trace [SEQ][B][HID] bf16
    const float*    __restrict__ Wlin,   // [HID]
    const float*    __restrict__ blin,   // [1]
    float*          __restrict__ out,    // [SEQ*B]
    int*            __restrict__ prog)   // cross-workgroup progress counter
{
  __shared__ uint16_t hbuf[BATCH * HPITCH];
  __shared__ float    outacc[BATCH];

  const int tid  = threadIdx.x;
  const int jt   = tid >> 5;    // wave id = hidden column block 0..7
  const int lane = tid & 31;
  const int half = lane >> 4;   // K-half selector inside fragments
  const int ln   = lane & 15;

  const int kb_a = half * 8;    // A per-lane K sub-offset
  const int nc   = jt * 16 + ln;// hidden column 0..127 (also W row % 128)
  const int bk0  = half * 8;    // B per-lane K sub-offset

  // Per-lane constants (invariant over t)
  float bgate[4];
#pragma unroll
  for (int g = 0; g < 4; ++g) bgate[g] = bias[g * HID + nc];
  const float wl = (LAYER == 1) ? Wlin[nc] : 0.f;
  const float bl = (LAYER == 1) ? blin[0] : 0.f;

  for (int i = tid; i < BATCH * HPITCH; i += 256) hbuf[i] = 0;
  if (LAYER == 1 && tid < BATCH) outacc[tid] = 0.f;

  v8f cst[4];  // cell state: 4 batch tiles x (16x16)/wave
#pragma unroll
  for (int mt = 0; mt < 4; ++mt)
#pragma unroll
    for (int v = 0; v < 8; ++v) cst[mt][v] = 0.f;

  __syncthreads();

  int zk = 0;  // opaque zero: keeps weight loads in-loop without flat-ifying them

#pragma unroll 1
  for (int t = 0; t < SEQ; ++t) {
    asm volatile("" : "+s"(zk));

    if (LAYER == 1) {
      // Wait until the producer has published h0[t] (progress >= t+1).
      while (__hip_atomic_load(prog, __ATOMIC_RELAXED, __HIP_MEMORY_SCOPE_AGENT) < t + 1)
        __builtin_amdgcn_s_sleep(1);
      __builtin_amdgcn_fence(__ATOMIC_ACQUIRE, "agent");  // invalidate stale lines
    }

    // Per-lane base pointers, computed ONCE per step; zk folds in here so all
    // fragment loads below use compile-time immediate offsets (no per-fragment
    // 64-bit address chains), yet remain un-hoistable.
    const uint16_t* WihL = Wih + nc * KIN + bk0 + zk;             // B: input weights
    const uint16_t* WhhL = Whh + nc * HID + bk0 + zk;             // B: recurrent weights
    const uint16_t* xL   = xbf + (size_t)t * BATCH * KIN + ln * KIN + kb_a;  // A: input

    v8f acc[4][4];  // [gate][mtile]
#pragma unroll
    for (int g = 0; g < 4; ++g)
#pragma unroll
      for (int mt = 0; mt < 4; ++mt)
#pragma unroll
        for (int v = 0; v < 8; ++v) acc[g][mt][v] = bgate[g];

    // ---- input projection: G += x_t @ Wih^T (no dependence on h_{t-1}) ----
    {
      // prefetch next step's input rows (HBM stream; producer layer only —
      // the consumer's input is freshly written and already L2-resident)
      if (LAYER == 0 && t + 1 < SEQ)
        __builtin_prefetch(xL + BATCH * KIN, 0, 1);
#pragma unroll
      for (int kc = 0; kc < KIN / 32; ++kc) {
        v16bf Af[4];
#pragma unroll
        for (int mt = 0; mt < 4; ++mt)
          Af[mt] = load_a_ofs(xL, mt * 16 * KIN + kc * 32);
#pragma unroll
        for (int g = 0; g < 4; ++g) {
          v16bf Bf = load_a_ofs(WihL, g * HID * KIN + kc * 32);
#pragma unroll
          for (int mt = 0; mt < 4; ++mt) acc[g][mt] = wmma_bf16(Af[mt], Bf, acc[g][mt]);
        }
      }
    }

    // ---- recurrent: G += h_{t-1} @ Whh^T (A from LDS, B from L2) ----
#pragma unroll
    for (int kc = 0; kc < HID / 32; ++kc) {
      v16bf Af[4];
#pragma unroll
      for (int mt = 0; mt < 4; ++mt) {
        FragU fa;
        const int m = mt * 16 + ln;
        fa.q[0] = *(const u32x4*)&hbuf[m * HPITCH + kc * 32 + kb_a];
        fa.q[1] = *(const u32x4*)&hbuf[m * HPITCH + kc * 32 + kb_a + 16];
        Af[mt] = fa.v;
      }
#pragma unroll
      for (int g = 0; g < 4; ++g) {
        v16bf Bf = load_a_ofs(WhhL, g * HID * HID + kc * 32);
#pragma unroll
        for (int mt = 0; mt < 4; ++mt) acc[g][mt] = wmma_bf16(Af[mt], Bf, acc[g][mt]);
      }
    }

    __syncthreads();  // everyone finished reading h_{t-1} from LDS

    // ---- LSTM cell elementwise; c never leaves VGPRs ----
#pragma unroll
    for (int mt = 0; mt < 4; ++mt) {
#pragma unroll
      for (int v = 0; v < 8; ++v) {
        float ig = sigm(acc[0][mt][v]);
        float fg = sigm(acc[1][mt][v]);
        float gg = tanh_(acc[2][mt][v]);
        float og = sigm(acc[3][mt][v]);
        float cn = fg * cst[mt][v] + ig * gg;
        cst[mt][v] = cn;
        float h = og * tanh_(cn);
        const int m = mt * 16 + v + 8 * half;  // C/D layout: VGPR v -> row v (+8 hi half)
        const uint16_t hb = f2bf(h);
        hbuf[m * HPITCH + nc] = hb;
        if (LAYER == 0) {
          hout[(size_t)t * BATCH * HID + (size_t)m * HID + nc] = hb;
        } else {
          // fused final linear: partial dot over this wave's 16 hidden columns
          float p = h * wl;
          p += __shfl_xor(p, 1);
          p += __shfl_xor(p, 2);
          p += __shfl_xor(p, 4);
          p += __shfl_xor(p, 8);
          if (ln == 0) atomicAdd(&outacc[m], p);
        }
      }
    }

    if (LAYER == 0) {
      // Make this thread's h-stores visible device-wide before publishing.
      __builtin_amdgcn_fence(__ATOMIC_RELEASE, "agent");
    }

    __syncthreads();  // h_t published in LDS; outacc complete; all fences done

    if (LAYER == 0) {
      if (tid == 0)
        __hip_atomic_store(prog, t + 1, __ATOMIC_RELEASE, __HIP_MEMORY_SCOPE_AGENT);
    } else {
      if (tid < BATCH) {
        out[(size_t)t * BATCH + tid] = outacc[tid] + bl;
        outacc[tid] = 0.f;  // ordered before next step's adds by the next barrier
      }
    }
  }
}

// Two-stage pipeline in a single launch: block 0 = layer 0 (producer),
// block 1 = layer 1 + fused linear (consumer), overlapped via progress flag.
__global__ __launch_bounds__(256) void lstm_pipeline_kernel(
    const uint16_t* __restrict__ xbf0, const uint16_t* __restrict__ h0,
    const uint16_t* __restrict__ wih0, const uint16_t* __restrict__ whh0,
    const uint16_t* __restrict__ wih1, const uint16_t* __restrict__ whh1,
    const float* __restrict__ b0, const float* __restrict__ b1,
    const float* __restrict__ Wlin, const float* __restrict__ blin,
    float* __restrict__ out, int* __restrict__ prog)
{
  if (blockIdx.x == 0) {
    lstm_layer_body<INDIM, 0>(xbf0, wih0, whh0, b0, (uint16_t*)h0, Wlin, blin, out, prog);
  } else {
    lstm_layer_body<HID, 1>(h0, wih1, whh1, b1, nullptr, Wlin, blin, out, prog);
  }
}

// ---------------------------------------------------------------------------
// One-shot prep: weights f32 -> bf16 (RNE), fold biases, pre-convert the whole
// input stream to bf16, and reset the pipeline progress flag (graph-replay safe).
// ---------------------------------------------------------------------------
__global__ void prep_kernel(const float* __restrict__ data,
                            const float* __restrict__ Wih0, const float* __restrict__ Whh0,
                            const float* __restrict__ bih0, const float* __restrict__ bhh0,
                            const float* __restrict__ Wih1, const float* __restrict__ Whh1,
                            const float* __restrict__ bih1, const float* __restrict__ bhh1,
                            uint16_t* __restrict__ xbf0,
                            uint16_t* __restrict__ wih0, uint16_t* __restrict__ whh0,
                            uint16_t* __restrict__ wih1, uint16_t* __restrict__ whh1,
                            float* __restrict__ b0, float* __restrict__ b1,
                            int* __restrict__ prog) {
  const int stride = gridDim.x * blockDim.x;
  const int i0 = blockIdx.x * blockDim.x + threadIdx.x;
  if (i0 == 0) *prog = 0;
  for (int i = i0; i < SEQ * BATCH * INDIM; i += stride) xbf0[i] = f2bf(data[i]);
  for (int i = i0; i < GATES * INDIM; i += stride) wih0[i] = f2bf(Wih0[i]);
  for (int i = i0; i < GATES * HID; i += stride) {
    whh0[i] = f2bf(Whh0[i]);
    wih1[i] = f2bf(Wih1[i]);
    whh1[i] = f2bf(Whh1[i]);
  }
  for (int i = i0; i < GATES; i += stride) {
    b0[i] = bih0[i] + bhh0[i];
    b1[i] = bih1[i] + bhh1[i];
  }
}

extern "C" void kernel_launch(void* const* d_in, const int* in_sizes, int n_in,
                              void* d_out, int out_size, void* d_ws, size_t ws_size,
                              hipStream_t stream) {
  const float* data = (const float*)d_in[0];
  const float* Wih0 = (const float*)d_in[1];
  const float* Whh0 = (const float*)d_in[2];
  const float* bih0 = (const float*)d_in[3];
  const float* bhh0 = (const float*)d_in[4];
  const float* Wih1 = (const float*)d_in[5];
  const float* Whh1 = (const float*)d_in[6];
  const float* bih1 = (const float*)d_in[7];
  const float* bhh1 = (const float*)d_in[8];
  const float* Wlin = (const float*)d_in[9];
  const float* blin = (const float*)d_in[10];

  // Workspace (~50 MB): bf16 weights + folded biases + bf16 input + h trace + flag.
  uint8_t* p = (uint8_t*)d_ws;
  uint16_t* wih0 = (uint16_t*)p; p += (size_t)GATES * INDIM * 2;
  uint16_t* whh0 = (uint16_t*)p; p += (size_t)GATES * HID * 2;
  uint16_t* wih1 = (uint16_t*)p; p += (size_t)GATES * HID * 2;
  uint16_t* whh1 = (uint16_t*)p; p += (size_t)GATES * HID * 2;
  float*    b0   = (float*)p;    p += (size_t)GATES * 4;
  float*    b1   = (float*)p;    p += (size_t)GATES * 4;
  uint16_t* xbf0 = (uint16_t*)p; p += (size_t)SEQ * BATCH * INDIM * 2;
  uint16_t* h0   = (uint16_t*)p; p += (size_t)SEQ * BATCH * HID * 2;
  int*      prog = (int*)p;      p += 64;

  prep_kernel<<<1024, 256, 0, stream>>>(data, Wih0, Whh0, bih0, bhh0,
                                        Wih1, Whh1, bih1, bhh1,
                                        xbf0, wih0, whh0, wih1, whh1, b0, b1, prog);

  lstm_pipeline_kernel<<<2, 256, 0, stream>>>(xbf0, h0, wih0, whh0, wih1, whh1,
                                              b0, b1, Wlin, blin, (float*)d_out, prog);
}
